// LSTMTagger_76802605187159
// MI455X (gfx1250) — matrix operand order repaired and checked by reading, not compiled, not run on hardware
//
#include <hip/hip_runtime.h>
#include <cstdint>
#include <cstddef>

#define T_LEN 8192
#define E_DIM 512
#define H_DIM 512
#define G4H   2048
#define NTAGS 64

typedef __attribute__((ext_vector_type(16))) __bf16 v16bf;
typedef __attribute__((ext_vector_type(8)))  float  v8f;
typedef __attribute__((ext_vector_type(4)))  unsigned int u32x4;
typedef __attribute__((ext_vector_type(8)))  int i32x8;
typedef __attribute__((ext_vector_type(4)))  int i32x4;

struct U16x16 { uint4 a, b; };   // 32 bytes == v16bf

__device__ __forceinline__ unsigned short f2bf(float f) {
  unsigned int u = __builtin_bit_cast(unsigned int, f);
  u += 0x7FFFu + ((u >> 16) & 1u);           // round-to-nearest-even
  return (unsigned short)(u >> 16);
}

__device__ __forceinline__ float sigmoidf_(float x) {
  return 1.0f / (1.0f + __expf(-x));
}

// Low 32 bits of a generic pointer to LDS == LDS byte offset (ISA 10.2 aperture rule).
__device__ __forceinline__ unsigned lds_off_of(const void* p) {
  return (unsigned)(unsigned long long)p;
}

// ---------------------------------------------------------------- TDM 2D tile load -> LDS
// Tensor: row-major [tensor_d1][tensor_d0] bf16, row stride = stride_d0 elements.
// Tile:   [tile_d1][tile_d0] starting at gptr (tile start, not tensor start).
// LDS:    rows padded: 64B data + 16B pad (pad_interval=16 DWORDs, pad_amount=4 DWORDs)
//         -> row stride 80B == ushort[40], matching the fragment loaders below.
__device__ __forceinline__ void tdm_load_tile_2d(
    unsigned lds_off, const unsigned short* gptr,
    int tensor_d0, int tensor_d1, int tile_d0, int tile_d1, int stride_d0)
{
  unsigned long long ga = (unsigned long long)gptr;
  u32x4 g0;
  g0[0] = 1u;                                               // count=1 (valid descriptor)
  g0[1] = lds_off;                                          // lds_addr
  g0[2] = (unsigned)(ga & 0xFFFFFFFFull);                   // global_addr[31:0]
  g0[3] = (unsigned)((ga >> 32) & 0x01FFFFFFull)            // global_addr[56:32]
        | (2u << 30);                                       // type=2 ("image")

  i32x8 g1;
  g1[0] = (1 << 16)      // data_size = 1 -> 2 bytes
        | (1 << 20)      // pad_enable
        | (3 << 22)      // pad_interval = 16 DWORDs (64B)
        | (3 << 25);     // pad_amount   = 4 DWORDs (16B)
  g1[1] = (tensor_d0 & 0xFFFF) << 16;                       // abar_addr=0 | tensor_dim0[15:0]
  g1[2] = ((tensor_d0 >> 16) & 0xFFFF) | ((tensor_d1 & 0xFFFF) << 16);
  g1[3] = ((tensor_d1 >> 16) & 0xFFFF) | ((tile_d0 & 0xFFFF) << 16);
  g1[4] = (tile_d1 & 0xFFFF);                               // tile_dim1 | tile_dim2=0
  g1[5] = stride_d0;                                        // tensor_dim0_stride[31:0]
  g1[6] = 0;                                                // stride hi | dim1_stride lo
  g1[7] = 0;
  i32x4 gz4 = {0, 0, 0, 0};                                 // groups 2/3: 2D tensor
  i32x8 gz8 = {0, 0, 0, 0, 0, 0, 0, 0};                     // trailing group (clang-23 6-arg form)
  __builtin_amdgcn_tensor_load_to_lds(g0, g1, gz4, gz4, gz8, 0);
}

// ---------------------------------------------------------------- cast f32 -> bf16
__global__ __launch_bounds__(256) void k_cast_bf16(const float* __restrict__ in,
                                                   unsigned short* __restrict__ out, int n) {
  int i = blockIdx.x * 256 + threadIdx.x;
  if (i < n) out[i] = f2bf(in[i]);
}

// ---------------------------------------------------------------- cast+transpose: [K][N] f32 -> [N][K] bf16  (K == 512)
__global__ __launch_bounds__(256) void k_cast_transpose_bf16(
    const float* __restrict__ in, unsigned short* __restrict__ out, int total, int N) {
  int i = blockIdx.x * 256 + threadIdx.x;     // i = n*K + k
  if (i >= total) return;
  int n = i >> 9;
  int k = i & 511;
  out[i] = f2bf(in[(size_t)k * N + n]);
}

// ---------------------------------------------------------------- embedding gather -> bf16
__global__ __launch_bounds__(256) void k_embed(const int* __restrict__ sent,
                                               const float* __restrict__ emb,
                                               unsigned short* __restrict__ out) {
  int i = blockIdx.x * 256 + threadIdx.x;           // i < T*E
  int t = i >> 9;                                   // E == 512
  int e = i & 511;
  out[i] = f2bf(emb[(size_t)sent[t] * E_DIM + e]);
}

// ---------------------------------------------------------------- enc_out = h_fwd + h_bwd (bf16)
__global__ __launch_bounds__(256) void k_add_cast(const float* __restrict__ a,
                                                  const float* __restrict__ b,
                                                  unsigned short* __restrict__ out, int n) {
  int i = blockIdx.x * 256 + threadIdx.x;
  if (i < n) out[i] = f2bf(a[i] + b[i]);
}

// ---------------------------------------------------------------- bf16 WMMA GEMM, TDM-staged, double-buffered
// C(MxN,f32) = A(MxK,bf16,row-major) * Bt(NxK,bf16,row-major == B transposed)
// Block: 256 threads = 8 waves; block tile 128x64; wave tile 32x32 (4 WMMA accs); K chunk 32.
__global__ __launch_bounds__(256) void k_gemm_bf16_wmma(
    const unsigned short* __restrict__ A, const unsigned short* __restrict__ Bt,
    float* __restrict__ C, int M, int N, int K)
{
  __shared__ unsigned short As[2][128][40];   // [buf][m][k], 80B row stride (TDM pad)
  __shared__ unsigned short Bs[2][64][40];    // [buf][n][k]
  const int tid  = threadIdx.x;
  const int lane = tid & 31;
  const int wave = tid >> 5;
  const int wm = wave & 3;                 // 4 waves along M
  const int wn = wave >> 2;                // 2 waves along N
  const int blockM = blockIdx.x * 128;
  const int blockN = blockIdx.y * 64;
  const int nk = K >> 5;

  v8f acc00 = {0,0,0,0,0,0,0,0};
  v8f acc01 = acc00, acc10 = acc00, acc11 = acc00;

  // prologue: chunk 0 into buffer 0 (one issuing thread; TDM ignores EXEC, 1 DMA per wave)
  if (tid == 0) {
    tdm_load_tile_2d(lds_off_of(&As[0][0][0]), A  + (size_t)blockM * K, K, M, 32, 128, K);
    tdm_load_tile_2d(lds_off_of(&Bs[0][0][0]), Bt + (size_t)blockN * K, K, N, 32, 64,  K);
  }

  for (int i = 0; i < nk; ++i) {
    const int buf = i & 1;
    if (tid == 0) {
      if (i + 1 < nk) {
        const int kt2 = (i + 1) << 5;
        tdm_load_tile_2d(lds_off_of(&As[buf ^ 1][0][0]), A  + (size_t)blockM * K + kt2, K, M, 32, 128, K);
        tdm_load_tile_2d(lds_off_of(&Bs[buf ^ 1][0][0]), Bt + (size_t)blockN * K + kt2, K, N, 32, 64,  K);
        __builtin_amdgcn_s_wait_tensorcnt((short)2);   // chunk i retired (in-order), i+1 in flight
      } else {
        __builtin_amdgcn_s_wait_tensorcnt((short)0);
      }
    }
    __syncthreads();                                    // chunk i visible to all waves

    // A fragments: lane<16 -> K {0..7,16..23}; lane>=16 -> K {8..15,24..31}
    const int rA  = wm * 32 + (lane & 15);
    const int kbA = (lane >> 4) << 3;      // 0 or 8
    U16x16 ua;
    ua.a = *(const uint4*)&As[buf][rA][kbA];
    ua.b = *(const uint4*)&As[buf][rA][kbA + 16];
    v16bf a0 = __builtin_bit_cast(v16bf, ua);
    ua.a = *(const uint4*)&As[buf][rA + 16][kbA];
    ua.b = *(const uint4*)&As[buf][rA + 16][kbA + 16];
    v16bf a1 = __builtin_bit_cast(v16bf, ua);

    // B fragments: lane<16 -> K {0..7,16..23}; lane>=16 -> K {8..15,24..31}
    const int cB  = wn * 32 + (lane & 15);
    const int kbB = (lane < 16) ? 0 : 8;
    U16x16 ub;
    ub.a = *(const uint4*)&Bs[buf][cB][kbB];
    ub.b = *(const uint4*)&Bs[buf][cB][kbB + 16];
    v16bf b0 = __builtin_bit_cast(v16bf, ub);
    ub.a = *(const uint4*)&Bs[buf][cB + 16][kbB];
    ub.b = *(const uint4*)&Bs[buf][cB + 16][kbB + 16];
    v16bf b1 = __builtin_bit_cast(v16bf, ub);

    acc00 = __builtin_amdgcn_wmma_f32_16x16x32_bf16(false, a0, false, b0, (short)0, acc00, false, false);
    acc01 = __builtin_amdgcn_wmma_f32_16x16x32_bf16(false, a0, false, b1, (short)0, acc01, false, false);
    acc10 = __builtin_amdgcn_wmma_f32_16x16x32_bf16(false, a1, false, b0, (short)0, acc10, false, false);
    acc11 = __builtin_amdgcn_wmma_f32_16x16x32_bf16(false, a1, false, b1, (short)0, acc11, false, false);
    __syncthreads();                                    // all reads of buf done before reuse
  }

  // C/D layout: vgpr r -> row r (lanes 0-15) / row r+8 (lanes 16-31); col = lane&15
  const int rowOff = (lane < 16) ? 0 : 8;
  const int colL   = lane & 15;
  const int mBase  = blockM + wm * 32;
  const int nBase  = blockN + wn * 32;
  #pragma unroll
  for (int r = 0; r < 8; ++r) {
    C[(size_t)(mBase + r + rowOff)      * N + (nBase + colL)     ] = acc00[r];
    C[(size_t)(mBase + r + rowOff)      * N + (nBase + 16 + colL)] = acc01[r];
    C[(size_t)(mBase + 16 + r + rowOff) * N + (nBase + colL)     ] = acc10[r];
    C[(size_t)(mBase + 16 + r + rowOff) * N + (nBase + 16 + colL)] = acc11[r];
  }
}

// ---------------------------------------------------------------- encoder scans (2 blocks: fwd/bwd)
// gates[t] = pre[t] + h@Whh + c@Wch ; state in LDS; weights L2-resident (192MB L2).
__global__ __launch_bounds__(1024) void k_lstm_scan_enc(
    const float* __restrict__ preF, const float* __restrict__ preB,
    const float* __restrict__ WhhF, const float* __restrict__ WchF,
    const float* __restrict__ WhhB, const float* __restrict__ WchB,
    const float* __restrict__ h0F, const float* __restrict__ c0F,
    const float* __restrict__ h0B, const float* __restrict__ c0B,
    float* __restrict__ outF, float* __restrict__ outB)
{
  __shared__ float h[H_DIM], c[H_DIM], gates[G4H];
  const int dir = blockIdx.x;
  const float* pre = dir ? preB : preF;
  const float* Whh = dir ? WhhB : WhhF;
  const float* Wch = dir ? WchB : WchF;
  const float* h0  = dir ? h0B  : h0F;
  const float* c0  = dir ? c0B  : c0F;
  float* out       = dir ? outB : outF;
  const int tid = threadIdx.x;

  if (tid < H_DIM) { h[tid] = h0[tid]; c[tid] = c0[tid]; }
  __syncthreads();

  for (int s = 0; s < T_LEN; ++s) {
    const int t = dir ? (T_LEN - 1 - s) : s;
    const float* p = pre + (size_t)t * G4H;
    if (s + 1 < T_LEN) {
      const int tn = dir ? (T_LEN - 2 - s) : (s + 1);
      __builtin_prefetch(pre + (size_t)tn * G4H + tid, 0, 0);
    }
    float g0 = p[tid];
    float g1 = p[tid + 1024];
    #pragma unroll 4
    for (int k = 0; k < H_DIM; ++k) {
      const float hk = h[k], ck = c[k];
      const float* wh = Whh + (size_t)k * G4H;
      const float* wc = Wch + (size_t)k * G4H;
      g0 = fmaf(hk, wh[tid],        g0);
      g1 = fmaf(hk, wh[tid + 1024], g1);
      g0 = fmaf(ck, wc[tid],        g0);
      g1 = fmaf(ck, wc[tid + 1024], g1);
    }
    gates[tid] = g0; gates[tid + 1024] = g1;
    __syncthreads();
    if (tid < H_DIM) {
      const float ig = sigmoidf_(gates[tid]);
      const float fg = sigmoidf_(gates[tid + 512]);
      const float gg = tanhf(gates[tid + 1024]);
      const float og = sigmoidf_(gates[tid + 1536]);
      const float c2 = fg * c[tid] + ig * gg;
      const float h2 = og * tanhf(c2);
      c[tid] = c2; h[tid] = h2;
      out[(size_t)t * H_DIM + tid] = h2;
    }
    __syncthreads();
  }
}

// ---------------------------------------------------------------- decoder scan (h,c,m states)
__global__ __launch_bounds__(1024) void k_lstm_scan_dec(
    const float* __restrict__ pre,
    const float* __restrict__ W1, const float* __restrict__ W2,
    const float* __restrict__ W3, const float* __restrict__ W4,
    const float* __restrict__ h0, const float* __restrict__ c0,
    const float* __restrict__ m0,
    unsigned short* __restrict__ outBF)
{
  __shared__ float h[H_DIM], c[H_DIM], m[H_DIM], gates[G4H];
  const int tid = threadIdx.x;
  if (tid < H_DIM) { h[tid] = h0[tid]; c[tid] = c0[tid]; m[tid] = m0[tid]; }
  __syncthreads();

  for (int t = 0; t < T_LEN; ++t) {
    const float* p = pre + (size_t)t * G4H;
    if (t + 1 < T_LEN) __builtin_prefetch(pre + (size_t)(t + 1) * G4H + tid, 0, 0);
    float g0 = p[tid];
    float g1 = p[tid + 1024];
    #pragma unroll 4
    for (int k = 0; k < H_DIM; ++k) {
      const float hk = h[k], ck = c[k], mk = m[k];
      const float* w1 = W1 + (size_t)k * G4H;
      const float* w2 = W2 + (size_t)k * G4H;
      const float* w3 = W3 + (size_t)k * G4H;
      g0 = fmaf(hk, w1[tid],        g0);
      g1 = fmaf(hk, w1[tid + 1024], g1);
      g0 = fmaf(ck, w2[tid],        g0);
      g1 = fmaf(ck, w2[tid + 1024], g1);
      g0 = fmaf(mk, w3[tid],        g0);
      g1 = fmaf(mk, w3[tid + 1024], g1);
    }
    gates[tid] = g0; gates[tid + 1024] = g1;
    __syncthreads();
    if (tid < H_DIM) {
      const float ig = sigmoidf_(gates[tid]);
      const float fg = sigmoidf_(gates[tid + 512]);
      const float gg = tanhf(gates[tid + 1024]);
      const float og = sigmoidf_(gates[tid + 1536]);
      const float c2 = fg * c[tid] + ig * gg;
      const float h2 = og * tanhf(c2);
      c[tid] = c2; h[tid] = h2;
      outBF[(size_t)t * H_DIM + tid] = f2bf(h2);
    }
    __syncthreads();                       // c fully updated
    if (tid < H_DIM) {
      float acc = 0.f;                     // m2 = tanh(c2 @ W4)
      #pragma unroll 4
      for (int k = 0; k < H_DIM; ++k)
        acc = fmaf(c[k], W4[(size_t)k * H_DIM + tid], acc);
      m[tid] = tanhf(acc);
    }
    __syncthreads();
  }
}

// ---------------------------------------------------------------- log_softmax over 64 tags
__global__ __launch_bounds__(256) void k_logsoftmax(const float* __restrict__ ts,
                                                    const float* __restrict__ bias,
                                                    float* __restrict__ out) {
  int r = blockIdx.x * 256 + threadIdx.x;
  if (r >= T_LEN) return;
  const float* x = ts + (size_t)r * NTAGS;
  float mx = -3.4e38f;
  #pragma unroll 8
  for (int j = 0; j < NTAGS; ++j) mx = fmaxf(mx, x[j] + bias[j]);
  float ssum = 0.f;
  #pragma unroll 8
  for (int j = 0; j < NTAGS; ++j) ssum += __expf(x[j] + bias[j] - mx);
  const float l = mx + __logf(ssum);
  float* o = out + (size_t)r * NTAGS;
  #pragma unroll 8
  for (int j = 0; j < NTAGS; ++j) o[j] = x[j] + bias[j] - l;
}

// ---------------------------------------------------------------- driver
extern "C" void kernel_launch(void* const* d_in, const int* in_sizes, int n_in,
                              void* d_out, int out_size, void* d_ws, size_t ws_size,
                              hipStream_t stream)
{
  const int*   sentence = (const int*)  d_in[0];
  const float* emb      = (const float*)d_in[1];
  const float* ef_Wih   = (const float*)d_in[2];
  const float* ef_Whh   = (const float*)d_in[3];
  const float* ef_Wch   = (const float*)d_in[4];
  const float* eb_Wih   = (const float*)d_in[5];
  const float* eb_Whh   = (const float*)d_in[6];
  const float* eb_Wch   = (const float*)d_in[7];
  const float* ef_h0    = (const float*)d_in[8];
  const float* ef_c0    = (const float*)d_in[9];
  const float* eb_h0    = (const float*)d_in[10];
  const float* eb_c0    = (const float*)d_in[11];
  const float* d_W0     = (const float*)d_in[12];
  const float* d_W1     = (const float*)d_in[13];
  const float* d_W2     = (const float*)d_in[14];
  const float* d_W3     = (const float*)d_in[15];
  const float* d_W4     = (const float*)d_in[16];
  const float* d_h0     = (const float*)d_in[17];
  const float* d_c0     = (const float*)d_in[18];
  const float* d_m0     = (const float*)d_in[19];
  const float* tag_W    = (const float*)d_in[20];
  const float* tag_b    = (const float*)d_in[21];
  float* out = (float*)d_out;

  char* ws = (char*)d_ws;
  size_t off = 0;
  auto alloc = [&](size_t bytes) -> char* {
    char* p = ws + off;
    off += (bytes + 255) & ~(size_t)255;
    return p;
  };
  unsigned short* embBF  = (unsigned short*)alloc((size_t)T_LEN * E_DIM * 2);
  unsigned short* wfT    = (unsigned short*)alloc((size_t)E_DIM * G4H * 2);    // [4H][E]
  unsigned short* wbT    = (unsigned short*)alloc((size_t)E_DIM * G4H * 2);
  unsigned short* wdT    = (unsigned short*)alloc((size_t)H_DIM * G4H * 2);
  unsigned short* wtagT  = (unsigned short*)alloc((size_t)H_DIM * NTAGS * 2);  // [64][512]
  unsigned short* encBF  = (unsigned short*)alloc((size_t)T_LEN * H_DIM * 2);
  unsigned short* decBF  = (unsigned short*)alloc((size_t)T_LEN * H_DIM * 2);
  float* preF = (float*)alloc((size_t)T_LEN * G4H * 4);
  float* preB = (float*)alloc((size_t)T_LEN * G4H * 4);
  float* hF   = (float*)alloc((size_t)T_LEN * H_DIM * 4);
  float* hB   = (float*)alloc((size_t)T_LEN * H_DIM * 4);
  // aliases (stream-ordered reuse; producers of preF/preB fully consumed first)
  float* preD  = preF;
  float* tagSp = preB;

  const dim3 b256(256);
  k_cast_transpose_bf16<<<dim3((E_DIM * G4H) / 256), b256, 0, stream>>>(ef_Wih, wfT, E_DIM * G4H, G4H);
  k_cast_transpose_bf16<<<dim3((E_DIM * G4H) / 256), b256, 0, stream>>>(eb_Wih, wbT, E_DIM * G4H, G4H);
  k_cast_transpose_bf16<<<dim3((E_DIM * G4H) / 256), b256, 0, stream>>>(d_W0,  wdT, E_DIM * G4H, G4H);
  k_cast_transpose_bf16<<<dim3((H_DIM * NTAGS) / 256), b256, 0, stream>>>(tag_W, wtagT, H_DIM * NTAGS, NTAGS);
  k_embed<<<dim3((T_LEN * E_DIM) / 256), b256, 0, stream>>>(sentence, emb, embBF);

  const dim3 gGemm(T_LEN / 128, G4H / 64);
  k_gemm_bf16_wmma<<<gGemm, b256, 0, stream>>>(embBF, wfT, preF, T_LEN, G4H, E_DIM);
  k_gemm_bf16_wmma<<<gGemm, b256, 0, stream>>>(embBF, wbT, preB, T_LEN, G4H, E_DIM);

  k_lstm_scan_enc<<<dim3(2), dim3(1024), 0, stream>>>(
      preF, preB, ef_Whh, ef_Wch, eb_Whh, eb_Wch,
      ef_h0, ef_c0, eb_h0, eb_c0, hF, hB);

  k_add_cast<<<dim3((T_LEN * H_DIM) / 256), b256, 0, stream>>>(hF, hB, encBF, T_LEN * H_DIM);

  k_gemm_bf16_wmma<<<gGemm, b256, 0, stream>>>(encBF, wdT, preD, T_LEN, G4H, H_DIM);

  k_lstm_scan_dec<<<dim3(1), dim3(1024), 0, stream>>>(
      preD, d_W1, d_W2, d_W3, d_W4, d_h0, d_c0, d_m0, decBF);

  const dim3 gTag(T_LEN / 128, NTAGS / 64);
  k_gemm_bf16_wmma<<<gTag, b256, 0, stream>>>(decBF, wtagT, tagSp, T_LEN, NTAGS, H_DIM);

  k_logsoftmax<<<dim3(T_LEN / 256), b256, 0, stream>>>(tagSp, tag_b, out);
}
